// CGCNN_82746839925240
// MI455X (gfx1250) — matrix operand order repaired and checked by reading, not compiled, not run on hardware
//
#include <hip/hip_runtime.h>
#include <math.h>

// ---------------- problem constants ----------------
#define NNODES 50000
#define NEDGES 800000LL
#define NF 92
#define EF 41
#define GF 180
#define ZP 256   // padded Z = 2*NF+EF = 225 -> 256 (8 k-chunks of 32)
#define NP 96    // padded NF = 92 -> 96 (6 n-tiles of 16)

// ---------------- CDNA5 vector types ----------------
typedef __attribute__((ext_vector_type(16))) __bf16 v16bf;
typedef __attribute__((ext_vector_type(8)))  float  v8f;
typedef __attribute__((ext_vector_type(4)))  float  v4f;

union AFrag { v16bf v; v4f f4[2]; };

static __device__ __forceinline__ unsigned short f32_to_bf16(float f) {
    unsigned int u = __builtin_bit_cast(unsigned int, f);
    unsigned int r = 0x7FFFu + ((u >> 16) & 1u);   // round-to-nearest-even
    return (unsigned short)((u + r) >> 16);
}

static __device__ __forceinline__ float silu_f(float v) {
    return v / (1.f + __expf(-v));
}

// ---------------- weight pre-conversion ----------------
// conv weights fp32 [225][92] row-major -> bf16 col-major padded [96 cols][256 K]
// fc1 weight  fp32 [92][92]  row-major -> bf16 col-major padded [96 cols][96 K]
__global__ __launch_bounds__(256) void convert_weights_kernel(
    const float* __restrict__ w1f, const float* __restrict__ w1s,
    const float* __restrict__ w2f, const float* __restrict__ w2s,
    const float* __restrict__ w3f, const float* __restrict__ w3s,
    const float* __restrict__ fc1w,
    unsigned short* __restrict__ wpad, unsigned short* __restrict__ fc1pad)
{
    const long long CONV_TOT = 6LL * NP * ZP;          // 147456
    long long i = (long long)blockIdx.x * blockDim.x + threadIdx.x;
    if (i < CONV_TOT) {
        int mat = (int)(i / (NP * ZP));
        int rem = (int)(i % (NP * ZP));
        int col = rem / ZP;
        int k   = rem % ZP;
        const float* src = (mat == 0) ? w1f : (mat == 1) ? w1s :
                           (mat == 2) ? w2f : (mat == 3) ? w2s :
                           (mat == 4) ? w3f : w3s;
        float v = 0.f;
        if (col < NF && k < (2 * NF + EF)) v = src[(long long)k * NF + col];
        wpad[i] = f32_to_bf16(v);
    } else {
        long long r = i - CONV_TOT;
        if (r < (long long)NP * NP) {
            int col = (int)(r / NP);
            int k   = (int)(r % NP);
            float v = 0.f;
            if (col < NF && k < NF) v = fc1w[(long long)k * NF + col];
            fc1pad[r] = f32_to_bf16(v);
        }
    }
}

// ---------------- misc ----------------
__global__ __launch_bounds__(256) void zero_kernel(float* __restrict__ p, long long n) {
    long long i = (long long)blockIdx.x * blockDim.x + threadIdx.x;
    if (i < n) p[i] = 0.f;
}

__global__ __launch_bounds__(256) void node_update_kernel(
    const float* __restrict__ hin, const float* __restrict__ agg,
    float* __restrict__ hout, long long n)
{
    long long i = (long long)blockIdx.x * blockDim.x + threadIdx.x;
    if (i < n) {
        float v = hin[i] + agg[i];
        hout[i] = silu_f(v);
    }
}

// ---------------- CGConv edge kernel (WMMA) ----------------
// block = 192 threads = 6 waves; handles 32 edges (two 16-row M-tiles).
// LDS: z tile [32 rows][256 K] bf16. Each wave owns one 16-column n-tile and
// computes BOTH gates for BOTH M-tiles, reusing each B fragment across the two
// M-tiles (4 WMMAs per B-pair load). sigmoid*softplus fused in-register, then
// scattered with global f32 atomics.
__global__ __launch_bounds__(192) void cgconv_edge_kernel(
    const float* __restrict__ xin, const long long* __restrict__ ei,
    const float* __restrict__ ea,
    const unsigned short* __restrict__ wf_pad, const unsigned short* __restrict__ ws_pad,
    const float* __restrict__ bfv, const float* __restrict__ bsv,
    float* __restrict__ agg)
{
    __shared__ unsigned short zsh[32 * ZP];   // 16 KB
    __shared__ int dsh[32];
    __shared__ int ssh[32];
    const int tid = threadIdx.x;
    const long long ebase = (long long)blockIdx.x * 32;

    if (tid < 32) {
        long long e = ebase + tid;
        int s = 0, d = 0;
        if (e < NEDGES) { s = (int)ei[e]; d = (int)ei[NEDGES + e]; }
        ssh[tid] = s; dsh[tid] = d;
    }
    __syncthreads();

    // gather z = [x[dst] | x[src] | edge_attr], bf16, zero-padded
    for (int f = tid; f < 32 * ZP; f += 192) {
        int row = f >> 8;
        int col = f & (ZP - 1);
        long long e = ebase + row;
        float v = 0.f;
        if (e < NEDGES) {
            if (col < NF)                 v = xin[(long long)dsh[row] * NF + col];
            else if (col < 2 * NF)        v = xin[(long long)ssh[row] * NF + (col - NF)];
            else if (col < 2 * NF + EF)   v = ea[e * EF + (col - 2 * NF)];
        }
        zsh[f] = f32_to_bf16(v);
    }
    __syncthreads();

    const int lane = tid & 31;
    const int wave = tid >> 5;                 // n-tile 0..5
    const int r    = lane & 15;                // A row within M-tile
    const int ncol = wave * 16 + (lane & 15);  // padded output column 0..95
    const int kbA  = (lane >> 4) << 4;         // A K-half: 0 or 16
    const int kbB  = (lane >> 4) << 4;         // B K-half: 0 or 16

    const unsigned short* bfp = wf_pad + (size_t)ncol * ZP;
    const unsigned short* bsp = ws_pad + (size_t)ncol * ZP;

    v8f cf0 = {0.f,0.f,0.f,0.f,0.f,0.f,0.f,0.f};
    v8f cs0 = cf0, cf1 = cf0, cs1 = cf0;
#pragma unroll
    for (int kc = 0; kc < 8; ++kc) {
        AFrag a0, a1;
        const unsigned short* z0 = zsh + r * ZP + kc * 32 + kbA;
        a0.f4[0] = *(const v4f*)(z0);
        a0.f4[1] = *(const v4f*)(z0 + 8);
        const unsigned short* z1 = z0 + 16 * ZP;
        a1.f4[0] = *(const v4f*)(z1);
        a1.f4[1] = *(const v4f*)(z1 + 8);
        v16bf bF = *(const v16bf*)(bfp + kc * 32 + kbB);
        v16bf bS = *(const v16bf*)(bsp + kc * 32 + kbB);
        cf0 = __builtin_amdgcn_wmma_f32_16x16x32_bf16(false, a0.v, false, bF,
                                                      (short)0, cf0, false, false);
        cs0 = __builtin_amdgcn_wmma_f32_16x16x32_bf16(false, a0.v, false, bS,
                                                      (short)0, cs0, false, false);
        cf1 = __builtin_amdgcn_wmma_f32_16x16x32_bf16(false, a1.v, false, bF,
                                                      (short)0, cf1, false, false);
        cs1 = __builtin_amdgcn_wmma_f32_16x16x32_bf16(false, a1.v, false, bS,
                                                      (short)0, cs1, false, false);
    }

    if (ncol < NF) {
        const float bfb = bfv[ncol];
        const float bsb = bsv[ncol];
        const int mbase = (lane >> 4) << 3;    // C layout: lanes>=16 hold M=8..15
#pragma unroll
        for (int mt = 0; mt < 2; ++mt) {
            const v8f& cf = mt ? cf1 : cf0;
            const v8f& cs = mt ? cs1 : cs0;
#pragma unroll
            for (int j = 0; j < 8; ++j) {
                int m = mt * 16 + mbase + j;
                long long e = ebase + m;
                if (e < NEDGES) {
                    float fv = cf[j] + bfb;
                    float sv = cs[j] + bsb;
                    float sig = 1.f / (1.f + __expf(-fv));
                    float sp  = (sv > 20.f) ? sv : log1pf(__expf(sv));
                    atomicAdd(agg + (long long)dsh[m] * NF + ncol, sig * sp);
                }
            }
        }
    }
}

// ---------------- fc1 (node-wise [50000,92]@[92,92], WMMA) ----------------
// 32 nodes per block, two M-tiles sharing each B fragment.
__global__ __launch_bounds__(192) void fc1_kernel(
    const float* __restrict__ hin, const unsigned short* __restrict__ fc1pad,
    const float* __restrict__ bias, float* __restrict__ hout)
{
    __shared__ unsigned short ash[32 * NP];    // 6 KB
    const int tid = threadIdx.x;
    const long long nbase = (long long)blockIdx.x * 32;

    for (int f = tid; f < 32 * NP; f += 192) {
        int row = f / NP;
        int col = f % NP;
        long long node = nbase + row;
        float v = 0.f;
        if (node < NNODES && col < NF) v = hin[node * NF + col];
        ash[f] = f32_to_bf16(v);
    }
    __syncthreads();

    const int lane = tid & 31;
    const int wave = tid >> 5;
    const int r    = lane & 15;
    const int ncol = wave * 16 + (lane & 15);
    const int kbA  = (lane >> 4) << 4;
    const int kbB  = (lane >> 4) << 4;

    const unsigned short* bp = fc1pad + (size_t)ncol * NP;

    v8f c0 = {0.f,0.f,0.f,0.f,0.f,0.f,0.f,0.f};
    v8f c1 = c0;
#pragma unroll
    for (int kc = 0; kc < 3; ++kc) {
        AFrag a0, a1;
        const unsigned short* z0 = ash + r * NP + kc * 32 + kbA;
        a0.f4[0] = *(const v4f*)(z0);
        a0.f4[1] = *(const v4f*)(z0 + 8);
        const unsigned short* z1 = z0 + 16 * NP;
        a1.f4[0] = *(const v4f*)(z1);
        a1.f4[1] = *(const v4f*)(z1 + 8);
        v16bf b = *(const v16bf*)(bp + kc * 32 + kbB);
        c0 = __builtin_amdgcn_wmma_f32_16x16x32_bf16(false, a0.v, false, b,
                                                     (short)0, c0, false, false);
        c1 = __builtin_amdgcn_wmma_f32_16x16x32_bf16(false, a1.v, false, b,
                                                     (short)0, c1, false, false);
    }

    if (ncol < NF) {
        const float bb = bias[ncol];
        const int mbase = (lane >> 4) << 3;
#pragma unroll
        for (int mt = 0; mt < 2; ++mt) {
            const v8f& c = mt ? c1 : c0;
#pragma unroll
            for (int j = 0; j < 8; ++j) {
                long long node = nbase + mt * 16 + mbase + j;
                if (node < NNODES) {
                    float v = c[j] + bb;
                    hout[node * NF + ncol] = silu_f(v);
                }
            }
        }
    }
}

// ---------------- global mean pool ----------------
__global__ __launch_bounds__(256) void pool_kernel(const float* __restrict__ h,
                                                   float* __restrict__ pooled)
{
    __shared__ float acc[NF];
    const int tid = threadIdx.x;
    if (tid < NF) acc[tid] = 0.f;
    __syncthreads();
    const long long total = (long long)NNODES * NF;
    for (long long i = (long long)blockIdx.x * blockDim.x + tid; i < total;
         i += (long long)gridDim.x * blockDim.x) {
        atomicAdd(&acc[(int)(i % NF)], h[i]);
    }
    __syncthreads();
    if (tid < NF) atomicAdd(&pooled[tid], acc[tid]);
}

__global__ __launch_bounds__(512) void finalize_pool_kernel(
    const float* __restrict__ pooled, const float* __restrict__ glob,
    float* __restrict__ g0)
{
    int i = threadIdx.x;
    if (i < NF) g0[i] = pooled[i] * (1.f / (float)NNODES);
    else if (i < NF + GF) g0[i] = glob[i - NF];
}

// ---------------- tiny dense layers (MLP head) ----------------
__global__ __launch_bounds__(256) void dense_kernel(
    const float* __restrict__ in, const float* __restrict__ W,
    const float* __restrict__ b, float* __restrict__ out,
    int K, int N, int act)
{
    int j = blockIdx.x * blockDim.x + threadIdx.x;
    if (j >= N) return;
    float s = b[j];
    for (int k = 0; k < K; ++k) s += in[k] * W[(size_t)k * N + j];
    if (act) s = silu_f(s);
    out[j] = s;
}

// ---------------- workspace layout ----------------
static constexpr size_t alup(size_t x) { return (x + 255) & ~(size_t)255; }
static constexpr size_t SZ_WPAD   = (size_t)6 * NP * ZP * sizeof(unsigned short);
static constexpr size_t SZ_FC1PAD = (size_t)NP * NP * sizeof(unsigned short);
static constexpr size_t SZ_NODE   = (size_t)NNODES * NF * sizeof(float);
static constexpr size_t OFF_WPAD   = 0;
static constexpr size_t OFF_FC1PAD = alup(OFF_WPAD + SZ_WPAD);
static constexpr size_t OFF_AGG    = alup(OFF_FC1PAD + SZ_FC1PAD);
static constexpr size_t OFF_HA     = alup(OFF_AGG + SZ_NODE);
static constexpr size_t OFF_HB     = alup(OFF_HA + SZ_NODE);
static constexpr size_t OFF_POOL   = alup(OFF_HB + SZ_NODE);
static constexpr size_t OFF_G0     = alup(OFF_POOL + NF * sizeof(float));
static constexpr size_t OFF_G1     = alup(OFF_G0 + 272 * sizeof(float));
static constexpr size_t OFF_G2     = alup(OFF_G1 + 1024 * sizeof(float));
static constexpr size_t OFF_G3     = alup(OFF_G2 + 512 * sizeof(float));
static constexpr size_t OFF_G4     = alup(OFF_G3 + 256 * sizeof(float));
static constexpr size_t OFF_G5     = alup(OFF_G4 + 128 * sizeof(float));

extern "C" void kernel_launch(void* const* d_in, const int* in_sizes, int n_in,
                              void* d_out, int out_size, void* d_ws, size_t ws_size,
                              hipStream_t stream) {
    (void)in_sizes; (void)n_in; (void)out_size; (void)ws_size;

    const float*      x    = (const float*)d_in[0];
    const long long*  ei   = (const long long*)d_in[1];
    const float*      ea   = (const float*)d_in[2];
    const float*      glob = (const float*)d_in[3];
    // d_in[4] = batch (all zeros, single graph) -> unused
    const float* cwf[3] = { (const float*)d_in[5],  (const float*)d_in[9],  (const float*)d_in[13] };
    const float* cbf[3] = { (const float*)d_in[6],  (const float*)d_in[10], (const float*)d_in[14] };
    const float* cws[3] = { (const float*)d_in[7],  (const float*)d_in[11], (const float*)d_in[15] };
    const float* cbs[3] = { (const float*)d_in[8],  (const float*)d_in[12], (const float*)d_in[16] };
    const float* fc1W = (const float*)d_in[17]; const float* fc1b = (const float*)d_in[18];
    const float* fc2W = (const float*)d_in[19]; const float* fc2b = (const float*)d_in[20];
    const float* fc3W = (const float*)d_in[21]; const float* fc3b = (const float*)d_in[22];
    const float* fc4W = (const float*)d_in[23]; const float* fc4b = (const float*)d_in[24];
    const float* fc5W = (const float*)d_in[25]; const float* fc5b = (const float*)d_in[26];
    const float* fc6W = (const float*)d_in[27]; const float* fc6b = (const float*)d_in[28];
    const float* fc7W = (const float*)d_in[29]; const float* fc7b = (const float*)d_in[30];

    char* ws = (char*)d_ws;
    unsigned short* wpad   = (unsigned short*)(ws + OFF_WPAD);
    unsigned short* fc1pad = (unsigned short*)(ws + OFF_FC1PAD);
    float* agg    = (float*)(ws + OFF_AGG);
    float* ha     = (float*)(ws + OFF_HA);
    float* hb     = (float*)(ws + OFF_HB);
    float* pooled = (float*)(ws + OFF_POOL);
    float* g0 = (float*)(ws + OFF_G0);
    float* g1 = (float*)(ws + OFF_G1);
    float* g2 = (float*)(ws + OFF_G2);
    float* g3 = (float*)(ws + OFF_G3);
    float* g4 = (float*)(ws + OFF_G4);
    float* g5 = (float*)(ws + OFF_G5);

    // 1) weights -> padded col-major bf16
    {
        long long tot = 6LL * NP * ZP + (long long)NP * NP;
        int blocks = (int)((tot + 255) / 256);
        convert_weights_kernel<<<blocks, 256, 0, stream>>>(
            cwf[0], cws[0], cwf[1], cws[1], cwf[2], cws[2], fc1W, wpad, fc1pad);
    }

    const long long nelem = (long long)NNODES * NF;
    const int eltBlocks  = (int)((nelem + 255) / 256);
    const int edgeBlocks = (int)((NEDGES + 31) / 32);   // 25000
    const int fcBlocks   = (NNODES + 31) / 32;          // 1563

    // 2) three CGConv layers (ping-pong h buffers)
    const float* hin = x;
    float* bufs[2] = { ha, hb };
    int cur = 0;
    for (int c = 0; c < 3; ++c) {
        zero_kernel<<<eltBlocks, 256, 0, stream>>>(agg, nelem);
        cgconv_edge_kernel<<<edgeBlocks, 192, 0, stream>>>(
            hin, ei, ea,
            wpad + (size_t)(2 * c + 0) * NP * ZP,
            wpad + (size_t)(2 * c + 1) * NP * ZP,
            cbf[c], cbs[c], agg);
        node_update_kernel<<<eltBlocks, 256, 0, stream>>>(hin, agg, bufs[cur], nelem);
        hin = bufs[cur];
        cur ^= 1;
    }

    // 3) fc1 applied twice (WMMA), ping-pong
    fc1_kernel<<<fcBlocks, 192, 0, stream>>>(hin, fc1pad, fc1b, bufs[cur]);
    hin = bufs[cur]; cur ^= 1;
    fc1_kernel<<<fcBlocks, 192, 0, stream>>>(hin, fc1pad, fc1b, bufs[cur]);
    hin = bufs[cur];

    // 4) mean pool + concat global attr
    zero_kernel<<<1, 256, 0, stream>>>(pooled, NF);
    pool_kernel<<<512, 256, 0, stream>>>(hin, pooled);
    finalize_pool_kernel<<<1, 512, 0, stream>>>(pooled, glob, g0);

    // 5) MLP head (tiny)
    dense_kernel<<<4, 256, 0, stream>>>(g0, fc2W, fc2b, g1, 272, 1024, 1);
    dense_kernel<<<2, 256, 0, stream>>>(g1, fc3W, fc3b, g2, 1024, 512, 1);
    dense_kernel<<<1, 256, 0, stream>>>(g2, fc4W, fc4b, g3, 512, 256, 1);
    dense_kernel<<<1, 256, 0, stream>>>(g3, fc5W, fc5b, g4, 256, 128, 1);
    dense_kernel<<<1, 256, 0, stream>>>(g4, fc6W, fc6b, g5, 128, 64, 1);
    dense_kernel<<<1, 256, 0, stream>>>(g5, fc7W, fc7b, (float*)d_out, 64, 1, 0);
}
